// EdgeNetworkLayer_75754633166953
// MI455X (gfx1250) — matrix operand order repaired and checked by reading, not compiled, not run on hardware
//
#include <hip/hip_runtime.h>
#include <hip/hip_bf16.h>

typedef __attribute__((ext_vector_type(16))) _Float16 v16h;
typedef __attribute__((ext_vector_type(8)))  float    v8f;

#define D_HID 64
#define DD    4096   // 64*64

// ---------------- kernel: zero the aggregation buffer ----------------
__global__ void k_zero(float* __restrict__ p, int n) {
    int i = blockIdx.x * blockDim.x + threadIdx.x;
    if (i < n) p[i] = 0.f;
}

// ---------------- kernel: W2 (64 x 4096 f32) -> f16, pre-swizzled as WMMA B operands
// B operand for v_wmma_f32_16x16x32_f16: 32(K) x 16(N) f16, lane = ((K>>3)&1)*16 + N,
// halves[h] = B[K = (h>>3)*16 + (lane>=16)*8 + (h&7), N = lane&15].
// Storage: W2p[ ((nt*2 + t)*32 + lane)*16 + h ], K_global = t*32 + (h>>3)*16 + (lane>=16)*8 + (h&7),
//          N_global = nt*16 + (lane&15),  nt in [0,256), t in {0,1}.
__global__ void k_prep_w2(const float* __restrict__ W2, _Float16* __restrict__ W2p) {
    int idx  = blockIdx.x * blockDim.x + threadIdx.x;   // 262144 total
    int h    = idx & 15;
    int lane = (idx >> 4) & 31;
    int t    = (idx >> 9) & 1;
    int nt   = idx >> 10;
    int K = t * 32 + (h >> 3) * 16 + ((lane >> 4) & 1) * 8 + (h & 7);
    int N = nt * 16 + (lane & 15);
    W2p[idx] = (_Float16)W2[K * DD + N];
}

// ---------------- kernel: act = relu(ef @ W1 + b1) -> f16, pre-swizzled as WMMA A operands
// A operand: 16(M) x 32(K) f16, lane%16 = M, K = (h>>3)*16 + (lane>=16)*8 + (h&7).
// Storage per 16-edge tile: actp[ ((tile*2 + t)*32 + lane)*16 + h ], K_global = t*32 + K.
__global__ void k_edge_mlp(const float* __restrict__ ef, const float* __restrict__ W1,
                           const float* __restrict__ b1, _Float16* __restrict__ actp,
                           int E, int Epad) {
    __shared__ float efs[4 * 16];
    __shared__ float W1s[16 * 64];
    int tid = threadIdx.x;
    int idx = blockIdx.x * 256 + tid;     // one thread per (edge, k); block = 4 edges
    int e = idx >> 6, k = idx & 63;
    int e0 = blockIdx.x * 4;
    for (int q = tid; q < 16 * 64; q += 256) W1s[q] = W1[q];
    if (tid < 64) {
        int ee = e0 + (tid >> 4), col = tid & 15;
        efs[tid] = (ee < E) ? ef[ee * 16 + col] : 0.f;
    }
    __syncthreads();
    int le = e - e0;
    float acc = b1[k];
    #pragma unroll
    for (int d = 0; d < 16; d++) acc += efs[le * 16 + d] * W1s[d * 64 + k];
    float v = (e < E) ? fmaxf(acc, 0.f) : 0.f;   // pad rows -> 0 so they contribute nothing
    // inverse swizzle for this (e, k)
    int tile = e >> 4, er = e & 15;
    int t  = k >> 5, kl = k & 31;
    int g  = kl >> 4, hh = (kl >> 3) & 1, o = kl & 7;
    int h  = g * 8 + o;
    int lane = er + hh * 16;
    actp[((((size_t)tile * 2 + t) * 32 + lane) << 4) + h] = (_Float16)v;
}

// 16-lane XOR-tree reduction within each half-wave via ds_swizzle (immediate patterns,
// group-of-32 mode: offset = 0x1F | xor_mask << 10). Halves (lanes 0-15 / 16-31) stay
// independent for all masks < 16.
__device__ __forceinline__ float xor_reduce16(float v) {
    v += __int_as_float(__builtin_amdgcn_ds_swizzle(__float_as_int(v), 0x041F)); // xor 1
    v += __int_as_float(__builtin_amdgcn_ds_swizzle(__float_as_int(v), 0x081F)); // xor 2
    v += __int_as_float(__builtin_amdgcn_ds_swizzle(__float_as_int(v), 0x101F)); // xor 4
    v += __int_as_float(__builtin_amdgcn_ds_swizzle(__float_as_int(v), 0x201F)); // xor 8
    return v;
}

// ---------------- main fused kernel: A = act@W2 + b2 (WMMA, f32 accum), fused per-edge
// matvec msg[e] = A_e @ h[src[e]], scatter-add into m[tgt[e]].
// One wave = 32 edges (two 16-edge A groups sharing every B tile => half the W2 traffic).
__global__ void __launch_bounds__(128)
k_edge_transform(const float* __restrict__ hmat,
                 const long long* __restrict__ src,
                 const long long* __restrict__ tgt,
                 const _Float16* __restrict__ actp,
                 const _Float16* __restrict__ W2p,
                 const float* __restrict__ b2,
                 float* __restrict__ m,
                 int E, int tiles32) {
    int wave = (blockIdx.x * blockDim.x + threadIdx.x) >> 5;
    int lane = threadIdx.x & 31;
    if (wave >= tiles32) return;             // whole-wave uniform: EXEC all-1 for WMMA
    int hi  = (lane >> 4) & 1;
    int nlo = lane & 15;

    // A operands (pre-swizzled), gathered h[src], and tgt ids for both 16-edge groups
    v16h a[2][2];
    float hwv[2][4][8];
    int   tg[2][8];
    #pragma unroll
    for (int g = 0; g < 2; g++) {
        const v16h* ap = reinterpret_cast<const v16h*>(actp + ((size_t)(wave * 2 + g)) * 1024);
        a[g][0] = ap[lane];
        a[g][1] = ap[32 + lane];
        int eg[8];
        #pragma unroll
        for (int r = 0; r < 8; r++) {
            int e = wave * 32 + g * 16 + r + 8 * hi;
            eg[r]   = (e < E) ? (int)src[e] : -1;
            tg[g][r] = (e < E) ? (int)tgt[e] : -1;
        }
        #pragma unroll
        for (int s = 0; s < 4; s++)
            #pragma unroll
            for (int r = 0; r < 8; r++)
                hwv[g][s][r] = (eg[r] >= 0) ? hmat[eg[r] * D_HID + s * 16 + nlo] : 0.f;
    }

    float partial[2][8];
    #pragma unroll
    for (int g = 0; g < 2; g++)
        #pragma unroll
        for (int r = 0; r < 8; r++) partial[g][r] = 0.f;

    const v16h* bp = reinterpret_cast<const v16h*>(W2p);
    for (int i = 0; i < 64; i++) {           // output channel i; n = i*64 + j
        #pragma unroll
        for (int s = 0; s < 4; s++) {        // j-subtile: j = s*16 + (lane&15)
            int nt = i * 4 + s;
            float b2v = b2[nt * 16 + nlo];   // bias folded into C (constant down columns)
            v16h b0  = bp[(size_t)(nt * 2 + 0) * 32 + lane];
            v16h b1v = bp[(size_t)(nt * 2 + 1) * 32 + lane];
            #pragma unroll
            for (int g = 0; g < 2; g++) {
                v8f c;
                #pragma unroll
                for (int q = 0; q < 8; q++) c[q] = b2v;
                c = __builtin_amdgcn_wmma_f32_16x16x32_f16(false, a[g][0], false, b0,  (short)0, c, false, false);
                c = __builtin_amdgcn_wmma_f32_16x16x32_f16(false, a[g][1], false, b1v, (short)0, c, false, false);
                #pragma unroll
                for (int r = 0; r < 8; r++) partial[g][r] += c[r] * hwv[g][s][r];
            }
        }
        // finished all 64 j for channel i: reduce over the 16 j-lanes, scatter-add
        #pragma unroll
        for (int g = 0; g < 2; g++)
            #pragma unroll
            for (int r = 0; r < 8; r++) {
                float red = xor_reduce16(partial[g][r]);
                partial[g][r] = 0.f;
                if (nlo == 0 && tg[g][r] >= 0)       // lane 0 -> e=..+r, lane 16 -> e=..+8+r
                    atomicAdd(&m[tg[g][r] * D_HID + i], red);
            }
    }
}

// ---------------- GRU cell (torch semantics), tiny: 0.25 GFLOP ----------------
__global__ void k_gru(const float* __restrict__ m, const float* __restrict__ hmat,
                      const float* __restrict__ W_ih, const float* __restrict__ W_hh,
                      const float* __restrict__ b_ih, const float* __restrict__ b_hh,
                      float* __restrict__ out, int N) {
    __shared__ float ms[4][64], hs[4][64];
    int tid = threadIdx.x;
    int ln = tid >> 6, c = tid & 63;
    int node = blockIdx.x * 4 + ln;
    if (node < N) { ms[ln][c] = m[node * 64 + c]; hs[ln][c] = hmat[node * 64 + c]; }
    __syncthreads();
    if (node >= N) return;
    float ir = b_ih[c], iz = b_ih[64 + c], inn = b_ih[128 + c];
    float hr = b_hh[c], hz = b_hh[64 + c], hn = b_hh[128 + c];
    const float* wr = W_ih + c * 64;
    const float* wz = W_ih + (64 + c) * 64;
    const float* wn = W_ih + (128 + c) * 64;
    const float* ur = W_hh + c * 64;
    const float* uz = W_hh + (64 + c) * 64;
    const float* un = W_hh + (128 + c) * 64;
    #pragma unroll 8
    for (int d = 0; d < 64; d++) {
        float md = ms[ln][d], hd = hs[ln][d];
        ir += md * wr[d]; iz += md * wz[d]; inn += md * wn[d];
        hr += hd * ur[d]; hz += hd * uz[d]; hn += hd * un[d];
    }
    float r = 1.f / (1.f + __expf(-(ir + hr)));
    float z = 1.f / (1.f + __expf(-(iz + hz)));
    float nn = tanhf(inn + r * hn);
    out[node * 64 + c] = (1.f - z) * nn + z * hs[ln][c];
}

extern "C" void kernel_launch(void* const* d_in, const int* in_sizes, int n_in,
                              void* d_out, int out_size, void* d_ws, size_t ws_size,
                              hipStream_t stream) {
    const float*     h    = (const float*)d_in[0];
    const long long* eidx = (const long long*)d_in[1];   // int64 (2, E)
    const float*     ef   = (const float*)d_in[2];
    const float*     W1   = (const float*)d_in[3];
    const float*     b1   = (const float*)d_in[4];
    const float*     W2   = (const float*)d_in[5];
    const float*     b2   = (const float*)d_in[6];
    const float*     W_ih = (const float*)d_in[7];
    const float*     W_hh = (const float*)d_in[8];
    const float*     b_ih = (const float*)d_in[9];
    const float*     b_hh = (const float*)d_in[10];

    int Nn      = in_sizes[0] / D_HID;       // 5000
    int E       = in_sizes[2] / 16;          // 50000
    int tiles32 = (E + 31) / 32;             // 1563 waves, 32 edges each
    int Epad    = tiles32 * 32;

    const long long* src = eidx;
    const long long* tgt = eidx + E;

    // workspace layout (all offsets 256B-aligned)
    char* ws = (char*)d_ws;
    _Float16* W2p  = (_Float16*)ws;                                   // 262144 f16 = 512 KB
    _Float16* actp = (_Float16*)(ws + 524288);                        // Epad*64 f16
    size_t act_bytes = (size_t)Epad * 64 * 2;
    size_t m_off = 524288 + ((act_bytes + 255) & ~(size_t)255);
    float* m = (float*)(ws + m_off);                                  // Nn*64 f32

    k_zero<<<(Nn * D_HID + 255) / 256, 256, 0, stream>>>(m, Nn * D_HID);
    k_prep_w2<<<262144 / 256, 256, 0, stream>>>(W2, W2p);
    k_edge_mlp<<<(Epad * 64) / 256, 256, 0, stream>>>(ef, W1, b1, actp, E, Epad);

    int waves_per_block = 4;                  // 128-thread blocks
    int blocks = (tiles32 + waves_per_block - 1) / waves_per_block;
    k_edge_transform<<<blocks, 32 * waves_per_block, 0, stream>>>(
        h, src, tgt, actp, W2p, b2, m, E, tiles32);

    k_gru<<<(Nn + 3) / 4, 256, 0, stream>>>(m, h, W_ih, W_hh, b_ih, b_hh, (float*)d_out, Nn);
}
